// MLPEdgeUpdator_9509057593724
// MI455X (gfx1250) — matrix operand order repaired and checked by reading, not compiled, not run on hardware
//
#include <hip/hip_runtime.h>

typedef __attribute__((ext_vector_type(16))) __bf16 v16bf;
typedef __attribute__((ext_vector_type(8)))  float  v8f;
typedef __attribute__((ext_vector_type(4)))  float  v4f;

#define DIM       128
#define KDIM      384
#define FRAG_HALF (12 * 8 * 32 * 16)   // 49152 bf16 per hi/lo half

// C[E,256] = [edge_feat | node_feat[src] | edge_query] @ [W_feat | W_query] + bias
// head = blockIdx.x & 1 selects W_feat (0) or W_query (1).
__global__ __launch_bounds__(256, 1)
void edge_mlp_wmma_kernel(const float* __restrict__ edge_feat,
                          const float* __restrict__ edge_query,
                          const float* __restrict__ node_feat,
                          const int*   __restrict__ src_idx,
                          const float* __restrict__ W_feat,
                          const float* __restrict__ b_feat,
                          const float* __restrict__ W_query,
                          const float* __restrict__ b_query,
                          float* __restrict__ out,
                          int E)
{
    extern __shared__ __bf16 lds[];
    __bf16* bHiLds = lds;
    __bf16* bLoLds = lds + FRAG_HALF;

    const int head      = blockIdx.x & 1;
    const float* Wm     = head ? W_query : W_feat;
    const float* biasP  = head ? b_query : b_feat;
    float* outp         = out + (size_t)head * (size_t)E * DIM;

    const int tid = threadIdx.x;

    // ---- Stage W -> bf16 hi/lo fragments in LDS, pre-swizzled to the
    // 16-bit B-matrix (32x16 KxN) wave32 lane layout:
    //   lane = ((k%32 / 8)&1)*16 + n%16 ; element j = (k%8) + 8*(k%32 / 16) ; tile = (k/32, n/16)
    for (int idx = tid; idx < KDIM * DIM; idx += 256) {
        int k = idx >> 7;          // row of W, 0..383
        int n = idx & (DIM - 1);   // col of W, 0..127
        float w  = Wm[idx];
        __bf16 h = (__bf16)w;
        __bf16 l = (__bf16)(w - (float)h);
        int kc = k >> 5, kr = k & 31;
        int ln = (((kr >> 3) & 1) << 4) | (n & 15);
        int pos = (((((kc << 3) | (n >> 4)) << 5) | ln) << 4) | ((kr & 7) | ((kr >> 4) << 3));
        bHiLds[pos] = h;
        bLoLds[pos] = l;
    }
    __syncthreads();

    const int wave = tid >> 5;
    const int lane = tid & 31;
    const int mrow = lane & 15;          // A-matrix M = lane % 16
    const int kk   = (lane >> 4) << 3;   // lane half K base: 0 or 8
    const int ncol = lane & 15;

    // loop-invariant bias values for this lane's output columns
    float bv[8];
#pragma unroll
    for (int nt = 0; nt < 8; ++nt) bv[nt] = biasP[(nt << 4) | ncol];

    const int mBlocks = (E + 127) >> 7;
    const int mStride = gridDim.x >> 1;

    for (int mb = (int)(blockIdx.x >> 1); mb < mBlocks; mb += mStride) {
        int row  = mb * 128 + wave * 16 + mrow;
        int rowc = row < E ? row : E - 1;        // clamp: keep EXEC all-1s
        long src = (long)src_idx[rowc];
        const float* a0p = edge_feat  + (size_t)rowc * DIM;
        const float* a1p = node_feat  + (size_t)src  * DIM;
        const float* a2p = edge_query + (size_t)rowc * DIM;

        // prefetch next stripe's rows (incl. the node_feat gather target)
        int nmb = mb + mStride;
        if (nmb < mBlocks) {
            int prow = nmb * 128 + wave * 16 + mrow;
            if (prow >= E) prow = E - 1;
            long psrc = (long)src_idx[prow];
            __builtin_prefetch(edge_feat  + (size_t)prow * DIM, 0, 0);
            __builtin_prefetch(edge_query + (size_t)prow * DIM, 0, 0);
            __builtin_prefetch(node_feat  + (size_t)psrc * DIM, 0, 0);
        }

        v8f acc[8] = {};

        // Keep this loop ROLLED: full unroll spills (>384 VGPRs + scratch
        // reloads observed). Rolled body: 16 A-floats + 1 B pair live.
#pragma unroll 1
        for (int kc = 0; kc < 12; ++kc) {
            const float* ap = (kc < 4) ? (a0p + kc * 32)
                            : (kc < 8) ? (a1p + (kc - 4) * 32)
                                       : (a2p + (kc - 8) * 32);
            // 16-bit A (16x32) layout: this lane holds K = kk..kk+7 and kk+16..kk+23
            v4f f0 = *(const v4f*)(ap + kk);
            v4f f1 = *(const v4f*)(ap + kk + 4);
            v4f f2 = *(const v4f*)(ap + kk + 16);
            v4f f3 = *(const v4f*)(ap + kk + 20);
            float af[16] = { f0.x, f0.y, f0.z, f0.w, f1.x, f1.y, f1.z, f1.w,
                             f2.x, f2.y, f2.z, f2.w, f3.x, f3.y, f3.z, f3.w };
            v16bf aHi, aLo;
#pragma unroll
            for (int j = 0; j < 16; ++j) {
                __bf16 h = (__bf16)af[j];
                aHi[j] = h;
                aLo[j] = (__bf16)(af[j] - (float)h);
            }
#pragma unroll
            for (int nt = 0; nt < 8; ++nt) {
                int off = ((((kc << 3) | nt) << 5) | lane) << 4;
                v16bf bH = *(const v16bf*)(bHiLds + off);
                v16bf bL = *(const v16bf*)(bLoLds + off);
                // bf16x3 fp32-emulation: aHi*bHi + aHi*bLo + aLo*bHi
                acc[nt] = __builtin_amdgcn_wmma_f32_16x16x32_bf16(
                    false, aHi, false, bH, (short)0, acc[nt], false, false);
                acc[nt] = __builtin_amdgcn_wmma_f32_16x16x32_bf16(
                    false, aHi, false, bL, (short)0, acc[nt], false, false);
                acc[nt] = __builtin_amdgcn_wmma_f32_16x16x32_bf16(
                    false, aLo, false, bH, (short)0, acc[nt], false, false);
            }
        }

        // 32-bit C/D 16x16 layout: VGPR r -> M = r + 8*(lane/16), N = lane%16
        int rBase = mb * 128 + wave * 16 + ((lane >> 4) << 3);
        float* op = outp + (size_t)rBase * DIM + ncol;
        if (rBase + 8 <= E) {
            // full block: unpredicated stores
#pragma unroll
            for (int nt = 0; nt < 8; ++nt)
#pragma unroll
                for (int r = 0; r < 8; ++r)
                    op[(size_t)r * DIM + (nt << 4)] = acc[nt][r] + bv[nt];
        } else {
            // tail block
#pragma unroll
            for (int nt = 0; nt < 8; ++nt)
#pragma unroll
                for (int r = 0; r < 8; ++r)
                    if (rBase + r < E)
                        op[(size_t)r * DIM + (nt << 4)] = acc[nt][r] + bv[nt];
        }
    }
}

extern "C" void kernel_launch(void* const* d_in, const int* in_sizes, int n_in,
                              void* d_out, int out_size, void* d_ws, size_t ws_size,
                              hipStream_t stream) {
    (void)n_in; (void)out_size; (void)d_ws; (void)ws_size;
    const float* edge_feat  = (const float*)d_in[0];
    const float* edge_query = (const float*)d_in[1];
    const float* node_feat  = (const float*)d_in[2];
    const int*   src_idx    = (const int*)  d_in[3];
    const float* W_feat     = (const float*)d_in[4];
    const float* b_feat     = (const float*)d_in[5];
    const float* W_query    = (const float*)d_in[6];
    const float* b_query    = (const float*)d_in[7];
    float* out = (float*)d_out;

    int E = in_sizes[0] / DIM;

    size_t shmem = (size_t)2 * FRAG_HALF * 2;  // 196608 bytes of LDS
    (void)hipFuncSetAttribute((const void*)edge_mlp_wmma_kernel,
                              hipFuncAttributeMaxDynamicSharedMemorySize,
                              (int)shmem);

    dim3 grid(1024), block(256);
    edge_mlp_wmma_kernel<<<grid, block, shmem, stream>>>(
        edge_feat, edge_query, node_feat, src_idx,
        W_feat, b_feat, W_query, b_query, out, E);
}
